// Net_link_train_2190433321518
// MI455X (gfx1250) — matrix operand order repaired and checked by reading, not compiled, not run on hardware
//
#include <hip/hip_runtime.h>
#include <hip/hip_bf16.h>

// ---------------------------------------------------------------------------
// Types for CDNA5 WMMA (wave32): bf16 16x16x32, f32 accumulate.
// ---------------------------------------------------------------------------
typedef __bf16 bf16_t;
typedef bf16_t v16bf __attribute__((ext_vector_type(16)));
typedef float  v8f   __attribute__((ext_vector_type(8)));

#define WT_STRIDE 144   // bf16 elements per LDS row (288B, 32B-aligned rows)

// ---------------------------------------------------------------------------
// Hardware float atomic add (CDNA5 GLOBAL_ATOMIC_ADD_F32, no return).
// Inline asm guarantees the single L2-side instruction instead of a CAS loop
// the compiler may emit for generic-pointer atomicrmw fadd.
// ---------------------------------------------------------------------------
__device__ __forceinline__ void atomic_add_f32(float* p, float v) {
  asm volatile("global_atomic_add_f32 %0, %1, off"
               :
               : "v"(p), "v"(v)
               : "memory");
}

// ---------------------------------------------------------------------------
// zero a float buffer (grid-stride)
// ---------------------------------------------------------------------------
__global__ void zero_f32_kernel(float* __restrict__ p, long n) {
  long i = (long)blockIdx.x * blockDim.x + threadIdx.x;
  long stride = (long)gridDim.x * blockDim.x;
  for (; i < n; i += stride) p[i] = 0.0f;
}

// ---------------------------------------------------------------------------
// degree over dst (unit edge weights)
// ---------------------------------------------------------------------------
__global__ void degree_kernel(const int* __restrict__ dst,
                              float* __restrict__ deg, int E) {
  int i = blockIdx.x * blockDim.x + threadIdx.x;
  if (i < E) atomic_add_f32(&deg[dst[i]], 1.0f);
}

// deg -> deg^{-1/2} in place (0 where deg == 0)
__global__ void dinv_kernel(float* __restrict__ deg, int n) {
  int i = blockIdx.x * blockDim.x + threadIdx.x;
  if (i < n) {
    float d = deg[i];
    deg[i] = (d > 0.0f) ? rsqrtf(d) : 0.0f;
  }
}

// ---------------------------------------------------------------------------
// Out[M,128] = act(A[M,128]) @ W[128,128]  via v_wmma_f32_16x16x32_bf16.
// Block = 256 threads (8 waves); each block owns a 128-row strip of M;
// wave w owns rows [m0+16w, m0+16w+16). W staged transposed in LDS as bf16.
// ---------------------------------------------------------------------------
__global__ __launch_bounds__(256) void gemm128_wmma_kernel(
    const float* __restrict__ A, const float* __restrict__ W,
    float* __restrict__ Out, int M, int applyRelu) {
  __shared__ __align__(32) bf16_t WT[128 * WT_STRIDE];

  const int t = threadIdx.x;
  // Stage W transposed: WT[n][k] = (bf16) W[k][n]  (coalesced global reads)
  for (int idx = t; idx < 128 * 128; idx += 256) {
    int k = idx >> 7, n = idx & 127;
    WT[n * WT_STRIDE + k] = (bf16_t)W[idx];
  }
  __syncthreads();

  const int lane = t & 31;
  const int wave = t >> 5;
  const int m0   = blockIdx.x * 128 + wave * 16;

  // A-fragment lane mapping (16-bit A 16x32, contiguous k; B uses the same order):
  //   lanes 0-15:  M=lane,    K in {0..15}
  //   lanes 16-31: M=lane-16, K in {16..31}
  const int mloc = lane & 15;
  int arow = m0 + mloc; if (arow >= M) arow = M - 1;   // clamp (stores guarded)
  const float* aptr = A + (long)arow * 128;
  const int akb = (lane < 16) ? 0 : 16;

  // B-fragment lane mapping (16-bit B 32x16):
  //   lanes 0-15 hold K=0..15 of column n=lane; lanes 16-31 hold K=16..31
  const int n   = lane & 15;
  const int bkb = (lane < 16) ? 0 : 16;

  v8f acc[8] = {};   // 8 N-tiles of 16 -> full N=128

#pragma unroll
  for (int kt = 0; kt < 4; ++kt) {           // K = 128 in steps of 32
    v16bf a;
#pragma unroll
    for (int i = 0; i < 16; ++i) {
      int k = kt * 32 + i + akb;
      float v = aptr[k];
      if (applyRelu) v = (v > 0.0f) ? v : 0.0f;
      a[i] = (bf16_t)v;
    }
#pragma unroll
    for (int nt = 0; nt < 8; ++nt) {
      const bf16_t* bp = &WT[(nt * 16 + n) * WT_STRIDE + kt * 32 + bkb];
      v16bf b = *(const v16bf*)bp;           // 32B aligned (row=288B, off=32B*j)
      acc[nt] = __builtin_amdgcn_wmma_f32_16x16x32_bf16(
          /*neg_a=*/false, a, /*neg_b=*/false, b,
          /*c_mod=*/(short)0, acc[nt], /*reuse_a=*/false, /*reuse_b=*/false);
    }
  }

  // C/D layout: lanes 0-15 -> M = 0..7 in acc[.][i]; lanes 16-31 -> M = 8..15
  const int mb = (lane < 16) ? 0 : 8;
#pragma unroll
  for (int nt = 0; nt < 8; ++nt) {
#pragma unroll
    for (int i = 0; i < 8; ++i) {
      int r = m0 + mb + i;
      if (r < M) Out[(long)r * 128 + nt * 16 + n] = acc[nt][i];
    }
  }
}

// ---------------------------------------------------------------------------
// out[dst] += h[src] * dinv[src]*dinv[dst]   (one wave-lane per 4 features)
// h/out rows are L2-resident (25.6 MB); 32 lanes * float4 = one 512B row.
// All atomics are single hardware global_atomic_add_f32 ops hitting L2.
// ---------------------------------------------------------------------------
__global__ void aggregate_kernel(const int* __restrict__ src,
                                 const int* __restrict__ dst,
                                 const float* __restrict__ dinv,
                                 const float* __restrict__ h,
                                 float* __restrict__ out, int E) {
  long gid = (long)blockIdx.x * blockDim.x + threadIdx.x;
  if (gid >= (long)E * 32) return;
  int e = (int)(gid >> 5);
  int l = (int)(gid & 31);
  int s = src[e], d = dst[e];
  float nrm = dinv[s] * dinv[d];
  float4 hv = *(const float4*)(h + (long)s * 128 + l * 4);
  float* op = out + (long)d * 128 + l * 4;
  atomic_add_f32(op + 0, hv.x * nrm);
  atomic_add_f32(op + 1, hv.y * nrm);
  atomic_add_f32(op + 2, hv.z * nrm);
  atomic_add_f32(op + 3, hv.w * nrm);
}

// ---------------------------------------------------------------------------
// decode: out[i,:2] = [z[e0] , z[e1]] @ Wlin   (wave per edge, shuffle reduce)
// ---------------------------------------------------------------------------
__global__ __launch_bounds__(256) void decode_kernel(
    const int* __restrict__ pos, const int* __restrict__ neg,
    const float* __restrict__ z, const float* __restrict__ Wlin,
    float* __restrict__ out, int Ep, int En) {
  __shared__ float wl[512];                 // Wlin [256,2]
  for (int i = threadIdx.x; i < 512; i += blockDim.x) wl[i] = Wlin[i];
  __syncthreads();

  long wid  = ((long)blockIdx.x * blockDim.x + threadIdx.x) >> 5;
  int  lane = threadIdx.x & 31;
  long Ed   = (long)Ep + En;
  if (wid >= Ed) return;

  int s, d;
  if (wid < Ep) { int i = (int)wid;        s = pos[i]; d = pos[Ep + i]; }
  else          { int i = (int)(wid - Ep); s = neg[i]; d = neg[En + i]; }

  int f = lane * 4;
  float4 a = *(const float4*)(z + (long)s * 128 + f);   // coalesced: 32x16B = row
  float4 b = *(const float4*)(z + (long)d * 128 + f);

  float p0, p1;
  p0  = a.x * wl[(f + 0) * 2 + 0] + a.y * wl[(f + 1) * 2 + 0]
      + a.z * wl[(f + 2) * 2 + 0] + a.w * wl[(f + 3) * 2 + 0];
  p1  = a.x * wl[(f + 0) * 2 + 1] + a.y * wl[(f + 1) * 2 + 1]
      + a.z * wl[(f + 2) * 2 + 1] + a.w * wl[(f + 3) * 2 + 1];
  p0 += b.x * wl[(128 + f + 0) * 2 + 0] + b.y * wl[(128 + f + 1) * 2 + 0]
      + b.z * wl[(128 + f + 2) * 2 + 0] + b.w * wl[(128 + f + 3) * 2 + 0];
  p1 += b.x * wl[(128 + f + 0) * 2 + 1] + b.y * wl[(128 + f + 1) * 2 + 1]
      + b.z * wl[(128 + f + 2) * 2 + 1] + b.w * wl[(128 + f + 3) * 2 + 1];

#pragma unroll
  for (int off = 16; off > 0; off >>= 1) {
    p0 += __shfl_down(p0, off);
    p1 += __shfl_down(p1, off);
  }
  if (lane == 0) {
    out[wid * 2 + 0] = p0;
    out[wid * 2 + 1] = p1;
  }
}

// ---------------------------------------------------------------------------
// driver
// ---------------------------------------------------------------------------
extern "C" void kernel_launch(void* const* d_in, const int* in_sizes, int n_in,
                              void* d_out, int out_size, void* d_ws, size_t ws_size,
                              hipStream_t stream) {
  const float* x    = (const float*)d_in[0];   // [N,128]
  const float* W1   = (const float*)d_in[1];   // [128,128]
  const float* W2   = (const float*)d_in[2];   // [128,128]
  const float* Wlin = (const float*)d_in[3];   // [256,2]
  const int*   ei   = (const int*)d_in[4];     // [2,E]
  const int*   pos  = (const int*)d_in[5];     // [2,Ep]
  const int*   neg  = (const int*)d_in[6];     // [2,En]

  const int H  = 128;
  const int N  = in_sizes[0] / H;   // 50000
  const int E  = in_sizes[4] / 2;   // 1.6M
  const int Ep = in_sizes[5] / 2;   // 500K
  const int En = in_sizes[6] / 2;   // 500K
  const int* src = ei;
  const int* dst = ei + E;

  // workspace layout (256B aligned chunks)
  auto aup = [](size_t v) { return (v + 255) & ~(size_t)255; };
  char* ws = (char*)d_ws;
  size_t off = 0;
  float* dinv = (float*)(ws + off); off += aup((size_t)N * 4);
  float* hbuf = (float*)(ws + off); off += aup((size_t)N * H * 4);
  float* z1   = (float*)(ws + off); off += aup((size_t)N * H * 4);
  float* z2   = (float*)(ws + off); off += aup((size_t)N * H * 4);
  (void)ws_size; (void)n_in; (void)out_size;

  const int T = 256;
  long NH = (long)N * H;

  // deg -> dinv
  zero_f32_kernel<<<(N + T - 1) / T, T, 0, stream>>>(dinv, N);
  degree_kernel<<<(E + T - 1) / T, T, 0, stream>>>(dst, dinv, E);
  dinv_kernel<<<(N + T - 1) / T, T, 0, stream>>>(dinv, N);

  // conv1: h = x @ W1 ; z1 = scatter-add(h[src]*norm)
  gemm128_wmma_kernel<<<(N + 127) / 128, T, 0, stream>>>(x, W1, hbuf, N, 0);
  zero_f32_kernel<<<4096, T, 0, stream>>>(z1, NH);
  aggregate_kernel<<<(int)(((long)E * 32 + T - 1) / T), T, 0, stream>>>(
      src, dst, dinv, hbuf, z1, E);

  // conv2: h = relu(z1) @ W2 ; z2 = scatter-add(h[src]*norm)
  gemm128_wmma_kernel<<<(N + 127) / 128, T, 0, stream>>>(z1, W2, hbuf, N, 1);
  zero_f32_kernel<<<4096, T, 0, stream>>>(z2, NH);
  aggregate_kernel<<<(int)(((long)E * 32 + T - 1) / T), T, 0, stream>>>(
      src, dst, dinv, hbuf, z2, E);

  // decode
  long Ed = (long)Ep + En;
  decode_kernel<<<(int)((Ed * 32 + T - 1) / T), T, 0, stream>>>(
      pos, neg, z2, Wlin, (float*)d_out, Ep, En);
}
